// latent2Transf_Decoder_79207786873357
// MI455X (gfx1250) — matrix operand order repaired and checked by reading, not compiled, not run on hardware
//
#include <hip/hip_runtime.h>
#include <hip/hip_bf16.h>

typedef __attribute__((ext_vector_type(16))) _Float16 v16h;
typedef __attribute__((ext_vector_type(8)))  _Float16 v8h;
typedef __attribute__((ext_vector_type(8)))  float    v8f;
typedef __attribute__((ext_vector_type(4)))  float    v4f;

#define BDIM        128        // 4 wave32s
#define WAVES       4
#define ROWS_TILE   64         // 4 waves * 16 rows
#define GRIDX       512
#define TILES_PB    4          // 512*4*64 = 131072 rows
#define NROWS       131072

__device__ __forceinline__ v8f wmma32(v16h a, v16h b, v8f c) {
    // D = A(16x32 f16) * B(32x16 f16) + C(16x16 f32)
    return __builtin_amdgcn_wmma_f32_16x16x32_f16(
        /*neg_a=*/false, a, /*neg_b=*/false, b,
        /*c_mod=*/(short)0, c, /*reuse_a=*/false, /*reuse_b=*/false);
}

// A-operand chunk (16x32 K-slice) from row-major f16 slab [16][128].
// ISA layout: lanes 0-15 (M=lane): j=0..7 -> K=kc*32+j, j=8..15 -> K=kc*32+16+(j-8)
//             lanes 16-31 shift K by +8.
__device__ __forceinline__ v16h load_a_chunk(const _Float16* hs, int r, int hi, int kc) {
    const int c1 = kc * 32 + hi * 8;
    v8h lo = *(const v8h*)(hs + r * 128 + c1);
    v8h hv = *(const v8h*)(hs + r * 128 + c1 + 16);
    return __builtin_shufflevector(lo, hv, 0,1,2,3,4,5,6,7,8,9,10,11,12,13,14,15);
}

__device__ __forceinline__ void mat3mul(const float* A, const float* B, float* C) {
#pragma unroll
    for (int i = 0; i < 3; ++i)
#pragma unroll
        for (int j = 0; j < 3; ++j)
            C[i*3+j] = fmaf(A[i*3+0], B[0+j],
                       fmaf(A[i*3+1], B[3+j],
                            A[i*3+2] * B[6+j]));
}

// expm of 3x3 via scaling-and-squaring + order-12 Taylor
__device__ __forceinline__ void expm3(const float* Ain, float* E) {
    float n0 = fabsf(Ain[0]) + fabsf(Ain[1]) + fabsf(Ain[2]);
    float n1 = fabsf(Ain[3]) + fabsf(Ain[4]) + fabsf(Ain[5]);
    float n2 = fabsf(Ain[6]) + fabsf(Ain[7]) + fabsf(Ain[8]);
    float nrm = fmaxf(n0, fmaxf(n1, n2));
    int s = 0;
    if (nrm > 0.25f) {
        s = (int)ceilf(log2f(nrm)) + 2;
        if (s < 0)  s = 0;
        if (s > 40) s = 40;
    }
    float sc = exp2f(-(float)s);
    float M[9], T[9], TM[9];
#pragma unroll
    for (int i = 0; i < 9; ++i) { M[i] = Ain[i] * sc; T[i] = M[i]; E[i] = M[i]; }
    E[0] += 1.0f; E[4] += 1.0f; E[8] += 1.0f;
#pragma unroll
    for (int k = 2; k <= 12; ++k) {
        mat3mul(T, M, TM);
        float inv = 1.0f / (float)k;
#pragma unroll
        for (int i = 0; i < 9; ++i) { T[i] = TM[i] * inv; E[i] += T[i]; }
    }
    for (int q = 0; q < s; ++q) {
        mat3mul(E, E, TM);
#pragma unroll
        for (int i = 0; i < 9; ++i) E[i] = TM[i];
    }
}

__global__ __launch_bounds__(BDIM)
void latent2transf_kernel(const float* __restrict__ y,
                          const float* __restrict__ ow0, const float* __restrict__ ob0,
                          const float* __restrict__ ow1, const float* __restrict__ ob1,
                          const float* __restrict__ ow2, const float* __restrict__ ob2,
                          const float* __restrict__ tw0, const float* __restrict__ tb0,
                          const float* __restrict__ tw1, const float* __restrict__ tb1,
                          const float* __restrict__ tw2, const float* __restrict__ tb2,
                          float* __restrict__ out) {
    // B-operand staging, pre-swizzled to WMMA layout: [kc][ntile][lane][16 halfs]
    __shared__ _Float16 wL0[4096];          // 32x128  (kc=1, n=8)
    __shared__ _Float16 wL1[16384];         // 128x128 (kc=4, n=8)
    __shared__ _Float16 wL2[2048];          // 128x16  (kc=4, n=1, zero-padded cols)
    __shared__ float    bL0[128], bL1[128], bL2[16];
    __shared__ _Float16 slab[WAVES * 2048]; // per-wave 16x128 f16 activation slab

    const int tid  = threadIdx.x;
    const int lane = tid & 31;
    const int wid  = tid >> 5;
    const int r    = lane & 15;   // M within tile
    const int hi   = lane >> 4;   // half-wave select
    const int nc   = lane & 15;   // N within tile (C/D layout)
    _Float16* hs   = slab + wid * 2048;

    for (int m = 0; m < 2; ++m) {
        const float* w0 = m ? tw0 : ow0;  const float* b0 = m ? tb0 : ob0;
        const float* w1 = m ? tw1 : ow1;  const float* b1 = m ? tb1 : ob1;
        const float* w2 = m ? tw2 : ow2;  const float* b2 = m ? tb2 : ob2;
        const int nout   = m ? 3 : 9;
        const int coloff = m ? 32 : 0;
        const int outoff = m ? 9 : 0;

        __syncthreads();
        // ---- cooperative weight staging (f32 -> f16, swizzled) ----
        for (int e = tid; e < 4096; e += BDIM) {        // L0: w0 is (128,32) [N][K]
            int j = e & 15, ln = (e >> 4) & 31, n = (e >> 9) & 7;
            int K = ((j >> 3) << 4) + ((ln >> 4) << 3) + (j & 7);
            int N = (n << 4) + (ln & 15);
            wL0[e] = (_Float16)w0[N * 32 + K];
        }
        for (int e = tid; e < 16384; e += BDIM) {       // L1: w1 is (128,128) [N][K]
            int j = e & 15, ln = (e >> 4) & 31, n = (e >> 9) & 7, kc = e >> 12;
            int K = kc * 32 + ((j >> 3) << 4) + ((ln >> 4) << 3) + (j & 7);
            int N = (n << 4) + (ln & 15);
            wL1[e] = (_Float16)w1[N * 128 + K];
        }
        for (int e = tid; e < 2048; e += BDIM) {        // L2: w2 is (nout,128), pad N to 16
            int j = e & 15, ln = (e >> 4) & 31, kc = e >> 9;
            int K = kc * 32 + ((j >> 3) << 4) + ((ln >> 4) << 3) + (j & 7);
            int N = ln & 15;
            wL2[e] = (N < nout) ? (_Float16)w2[N * 128 + K] : (_Float16)0.0f;
        }
        if (tid < 128) { bL0[tid] = b0[tid]; bL1[tid] = b1[tid]; }
        if (tid < 16)  { bL2[tid] = (tid < nout) ? b2[tid] : 0.0f; }
        __syncthreads();

        for (int t = 0; t < TILES_PB; ++t) {
            const int tile    = blockIdx.x * TILES_PB + t;
            const int rowbase = tile * ROWS_TILE + wid * 16;

            if (t + 1 < TILES_PB) {  // prefetch next tile's y rows
                const float* pf = y + (size_t)(tile + 1) * ROWS_TILE * 64;
                __builtin_prefetch(pf + tid * 32, 0, 1);
            }

            // ---------- layer 0: x(16x32) -> h0(16x128) ----------
            v16h a0;
            {
                const float* yr = y + (size_t)(rowbase + r) * 64 + coloff;
                v4f f0 = *(const v4f*)(yr + hi * 8);
                v4f f1 = *(const v4f*)(yr + hi * 8 + 4);
                v4f f2 = *(const v4f*)(yr + 16 + hi * 8);
                v4f f3 = *(const v4f*)(yr + 16 + hi * 8 + 4);
#pragma unroll
                for (int i = 0; i < 4; ++i) {
                    a0[i]      = (_Float16)f0[i];
                    a0[4 + i]  = (_Float16)f1[i];
                    a0[8 + i]  = (_Float16)f2[i];
                    a0[12 + i] = (_Float16)f3[i];
                }
            }
#pragma unroll
            for (int n = 0; n < 8; ++n) {
                float bv = bL0[n * 16 + nc];
                v8f c; 
#pragma unroll
                for (int v = 0; v < 8; ++v) c[v] = bv;
                v16h b = *(const v16h*)(wL0 + (n * 32 + lane) * 16);
                c = wmma32(a0, b, c);
#pragma unroll
                for (int v = 0; v < 8; ++v) {
                    float x = c[v] > 0.0f ? c[v] : 0.0f;     // relu
                    hs[(v + hi * 8) * 128 + n * 16 + nc] = (_Float16)x;
                }
            }
            __builtin_amdgcn_wave_barrier();

            // ---------- layer 1: h0(16x128) -> h1(16x128) ----------
            v16h A[4];
#pragma unroll
            for (int kc = 0; kc < 4; ++kc) A[kc] = load_a_chunk(hs, r, hi, kc);
            __builtin_amdgcn_wave_barrier();
#pragma unroll
            for (int n = 0; n < 8; ++n) {
                float bv = bL1[n * 16 + nc];
                v8f c;
#pragma unroll
                for (int v = 0; v < 8; ++v) c[v] = bv;
#pragma unroll
                for (int kc = 0; kc < 4; ++kc) {
                    v16h b = *(const v16h*)(wL1 + ((kc * 8 + n) * 32 + lane) * 16);
                    c = wmma32(A[kc], b, c);
                }
#pragma unroll
                for (int v = 0; v < 8; ++v) {
                    float x = c[v] > 0.0f ? c[v] : 0.0f;     // relu
                    hs[(v + hi * 8) * 128 + n * 16 + nc] = (_Float16)x;  // overwrite slab
                }
            }
            __builtin_amdgcn_wave_barrier();

            // ---------- layer 2: h1(16x128) -> out(16x16, nout valid) ----------
#pragma unroll
            for (int kc = 0; kc < 4; ++kc) A[kc] = load_a_chunk(hs, r, hi, kc);
            __builtin_amdgcn_wave_barrier();
            v8f c;
            {
                float bv = bL2[nc];
#pragma unroll
                for (int v = 0; v < 8; ++v) c[v] = bv;
            }
#pragma unroll
            for (int kc = 0; kc < 4; ++kc) {
                v16h b = *(const v16h*)(wL2 + (kc * 32 + lane) * 16);
                c = wmma32(A[kc], b, c);
            }

            if (m == 0) {
                // stage 16x16 f32 into slab, then one row-per-lane expm
                float* sf = (float*)hs;
#pragma unroll
                for (int v = 0; v < 8; ++v) sf[(v + hi * 8) * 16 + nc] = c[v];
                __builtin_amdgcn_wave_barrier();
                if (lane < 16) {
                    float Mw[9], E[9];
#pragma unroll
                    for (int i = 0; i < 9; ++i) Mw[i] = sf[lane * 16 + i];
                    expm3(Mw, E);
                    float* op = out + (size_t)(rowbase + lane) * 12;   // 48B-aligned
                    v4f e0 = {E[0], E[1], E[2], E[3]};
                    v4f e1 = {E[4], E[5], E[6], E[7]};
                    *(v4f*)(op)     = e0;
                    *(v4f*)(op + 4) = e1;
                    op[8] = E[8];
                }
                __builtin_amdgcn_wave_barrier();
            } else {
                if (nc < 3) {
#pragma unroll
                    for (int v = 0; v < 8; ++v)
                        out[(size_t)(rowbase + v + hi * 8) * 12 + outoff + nc] = c[v];
                }
            }
        }
    }
}

extern "C" void kernel_launch(void* const* d_in, const int* in_sizes, int n_in,
                              void* d_out, int out_size, void* d_ws, size_t ws_size,
                              hipStream_t stream) {
    (void)in_sizes; (void)n_in; (void)out_size; (void)d_ws; (void)ws_size;
    const float* y   = (const float*)d_in[0];
    const float* ow0 = (const float*)d_in[1];
    const float* ob0 = (const float*)d_in[2];
    const float* ow1 = (const float*)d_in[3];
    const float* ob1 = (const float*)d_in[4];
    const float* ow2 = (const float*)d_in[5];
    const float* ob2 = (const float*)d_in[6];
    const float* tw0 = (const float*)d_in[7];
    const float* tb0 = (const float*)d_in[8];
    const float* tw1 = (const float*)d_in[9];
    const float* tb1 = (const float*)d_in[10];
    const float* tw2 = (const float*)d_in[11];
    const float* tb2 = (const float*)d_in[12];
    float* out = (float*)d_out;

    latent2transf_kernel<<<dim3(GRIDX), dim3(BDIM), 0, stream>>>(
        y, ow0, ob0, ow1, ob1, ow2, ob2, tw0, tb0, tw1, tb1, tw2, tb2, out);
}
